// SelfAttention_1984274891264
// MI455X (gfx1250) — matrix operand order repaired and checked by reading, not compiled, not run on hardware
//
#include <hip/hip_runtime.h>
#include <hip/hip_bf16.h>

// Problem constants
#define NB 2
#define NT 2048
#define DM 1024
#define NH 16
#define DH 64
#define BT (NB * NT)   // 4096 rows

typedef __bf16 bf16;
typedef __attribute__((ext_vector_type(16))) __bf16 v16bf;
typedef __attribute__((ext_vector_type(8)))  __bf16 v8bf;
typedef __attribute__((ext_vector_type(8)))  float  v8f;

union AFrag { v16bf v; v8bf h[2]; };

__device__ inline v8f zero8() {
  v8f z;
#pragma unroll
  for (int i = 0; i < 8; ++i) z[i] = 0.0f;
  return z;
}

__device__ inline v8f wmma_bf16(v16bf a, v16bf b, v8f c) {
  // D = A(16x32 bf16) * B(32x16 bf16) + C(16x16 f32)
  return __builtin_amdgcn_wmma_f32_16x16x32_bf16(false, a, false, b,
                                                 (short)0, c, false, false);
}

// ---------------------------------------------------------------------------
// Kernel 1: fp32 -> bf16 casts; weights stored transposed (N-major, K contig)
// ---------------------------------------------------------------------------
__global__ void prep_kernel(const float* __restrict__ x,
                            const float* __restrict__ wqkv,
                            const float* __restrict__ wo,
                            bf16* __restrict__ xb,
                            bf16* __restrict__ wqkvt,   // [3072][1024]
                            bf16* __restrict__ wot) {   // [1024][1024]
  const int NX  = BT * DM;       // 4194304
  const int NW1 = DM * 3 * DM;   // 3145728
  const int NW2 = DM * DM;       // 1048576
  const int total = NX + NW1 + NW2;
  for (int i = blockIdx.x * blockDim.x + threadIdx.x; i < total;
       i += gridDim.x * blockDim.x) {
    if (i < NX) {
      xb[i] = (bf16)x[i];
    } else if (i < NX + NW1) {
      int j = i - NX;
      int col = j / DM;          // output column 0..3071
      int k   = j % DM;
      wqkvt[j] = (bf16)wqkv[k * (3 * DM) + col];
    } else {
      int j = i - NX - NW1;
      int col = j / DM;
      int k   = j % DM;
      wot[j] = (bf16)wo[k * DM + col];
    }
  }
}

// ---------------------------------------------------------------------------
// Shared GEMM core: one wave computes C[32 x 64] = A[32 x K] * Bt[64 x K]^T.
// Two A fragments (row tiles) share four B fragments: 8 WMMA / 12 loads per
// 32-wide k-chunk. Double-buffered (unroll x2): while WMMAs consume one
// fragment set, all 12 loads of the other set are in flight.
// Kdim must be a power of two and a multiple of 64.
// ---------------------------------------------------------------------------
struct Frags { AFrag a0, a1; v16bf b[4]; };

__device__ inline void load_frags(const bf16* __restrict__ ap0,
                                  const bf16* __restrict__ ap1,
                                  const bf16* const bp[4],
                                  int k0, int hi, Frags& f) {
  f.a0.h[0] = *(const v8bf*)(ap0 + k0 + hi * 16);       // K 0-7 / 16-23
  f.a0.h[1] = *(const v8bf*)(ap0 + k0 + hi * 16 + 8);   // K 8-15 / 24-31
  f.a1.h[0] = *(const v8bf*)(ap1 + k0 + hi * 16);
  f.a1.h[1] = *(const v8bf*)(ap1 + k0 + hi * 16 + 8);
#pragma unroll
  for (int nb = 0; nb < 4; ++nb)
    f.b[nb] = *(const v16bf*)(bp[nb] + k0);             // 16 contiguous K vals
}

__device__ inline void frag_compute(const Frags& f, v8f acc[2][4]) {
#pragma unroll
  for (int nb = 0; nb < 4; ++nb) {
    acc[0][nb] = wmma_bf16(f.a0.v, f.b[nb], acc[0][nb]);
    acc[1][nb] = wmma_bf16(f.a1.v, f.b[nb], acc[1][nb]);
  }
}

__device__ inline void gemm_tile_32x64(const bf16* __restrict__ A,
                                       const bf16* __restrict__ Bt,
                                       int lda, int ldb, int Kdim,
                                       int mbase, int nbase, int lane,
                                       v8f acc[2][4]) {
  const int row = lane & 15;
  const int hi  = lane >> 4;
  const bf16* ap0 = A + (size_t)(mbase + row) * lda;
  const bf16* ap1 = ap0 + (size_t)16 * lda;
  const bf16* bp[4];
#pragma unroll
  for (int nb = 0; nb < 4; ++nb)
    bp[nb] = Bt + (size_t)(nbase + nb * 16 + row) * ldb + hi * 16;

  Frags f0, f1;
  load_frags(ap0, ap1, bp, 0, hi, f0);
  for (int k0 = 0; k0 < Kdim; k0 += 64) {
    load_frags(ap0, ap1, bp, k0 + 32, hi, f1);
    frag_compute(f0, acc);
    // wrap-masked: branchless; final iteration redundantly reloads chunk 0
    load_frags(ap0, ap1, bp, (k0 + 64) & (Kdim - 1), hi, f0);
    frag_compute(f1, acc);
  }
}

// ---------------------------------------------------------------------------
// Kernel 2: qkv = x @ W_qkv, scattered into Q(b,h,t,d), K(b,h,t,d), V^T(b,h,d,t)
// ---------------------------------------------------------------------------
__global__ void qkv_gemm_kernel(const bf16* __restrict__ xb,
                                const bf16* __restrict__ wqkvt,
                                bf16* __restrict__ qbuf,
                                bf16* __restrict__ kbuf,
                                bf16* __restrict__ vtbuf) {
  const int lane = threadIdx.x & 31;
  const int wave = threadIdx.x >> 5;
  const int wid  = blockIdx.x * (blockDim.x >> 5) + wave;
  const int NG   = (3 * DM) / 64;        // 48 column groups
  const int mt = wid / NG, ng = wid % NG;
  const int mbase = mt * 32, nbase = ng * 64;

  v8f acc[2][4];
#pragma unroll
  for (int q = 0; q < 2; ++q)
#pragma unroll
    for (int nb = 0; nb < 4; ++nb) acc[q][nb] = zero8();

  gemm_tile_32x64(xb, wqkvt, DM, DM, DM, mbase, nbase, lane, acc);

  const int hi = lane >> 4, n = lane & 15;
#pragma unroll
  for (int q = 0; q < 2; ++q) {
#pragma unroll
    for (int nb = 0; nb < 4; ++nb) {
#pragma unroll
      for (int r = 0; r < 8; ++r) {
        int m = mbase + q * 16 + r + 8 * hi;   // global row
        int b = m / NT, t = m % NT;
        int j = nbase + nb * 16 + n;           // global column 0..3071
        int type = j / DM, rem = j % DM;
        int h = rem / DH, d = rem % DH;
        bf16 val = (bf16)acc[q][nb][r];
        if (type == 0)
          qbuf[(size_t)((b * NH + h) * NT + t) * DH + d] = val;
        else if (type == 1)
          kbuf[(size_t)((b * NH + h) * NT + t) * DH + d] = val;
        else
          vtbuf[((size_t)(b * NH + h) * DH + d) * NT + t] = val;
      }
    }
  }
}

// ---------------------------------------------------------------------------
// Kernel 3: causal flash attention. One wave per 32-query strip (two 16-row
// q-tiles) so every K/V fragment load feeds two WMMA pairs.
// ---------------------------------------------------------------------------
__global__ void flash_kernel(const bf16* __restrict__ qbuf,
                             const bf16* __restrict__ kbuf,
                             const bf16* __restrict__ vtbuf,
                             bf16* __restrict__ attn) {  // [BT][DM]
  __shared__ __align__(32) bf16 plds[4][2][16 * 32];     // per-wave P tiles
  const int lane = threadIdx.x & 31;
  const int wave = threadIdx.x >> 5;
  const int wid  = blockIdx.x * 4 + wave;
  const int QT   = NT / 32;                              // 64 strips per head
  const int bh   = wid / QT;
  const int t0   = (wid % QT) * 32;

  const bf16* K  = kbuf  + (size_t)bh * NT * DH;
  const bf16* Vt = vtbuf + (size_t)bh * DH * NT;

  const int row = lane & 15;   // A-row / B-col / C-col index
  const int hi  = lane >> 4;
  const float scale = 0.125f;  // 1/sqrt(64)

  // Q fragments: [qtile][d-chunk], kept in registers throughout
  AFrag qf[2][2];
#pragma unroll
  for (int qt = 0; qt < 2; ++qt) {
    const bf16* Q = qbuf + (size_t)(bh * NT + t0 + qt * 16) * DH;
#pragma unroll
    for (int c = 0; c < 2; ++c) {
      const bf16* qp = Q + (size_t)row * DH + c * 32;
      qf[qt][c].h[0] = *(const v8bf*)(qp + hi * 16);
      qf[qt][c].h[1] = *(const v8bf*)(qp + hi * 16 + 8);
    }
  }

  float mrow[2][8], lrow[2][8];
  v8f o[2][4];
#pragma unroll
  for (int qt = 0; qt < 2; ++qt) {
#pragma unroll
    for (int r = 0; r < 8; ++r) { mrow[qt][r] = -1e30f; lrow[qt][r] = 0.0f; }
#pragma unroll
    for (int c = 0; c < 4; ++c) o[qt][c] = zero8();
  }

  for (int kb0 = 0; kb0 <= t0 + 31; kb0 += 32) {
    // ---- K fragments for this key block, shared by both q-tiles ----
    v16bf kf[2][2];   // [key half][d-chunk]
#pragma unroll
    for (int c = 0; c < 2; ++c) {
      kf[0][c] = *(const v16bf*)(K + (size_t)(kb0 + row) * DH + c * 32 + hi * 16);
      kf[1][c] = *(const v16bf*)(K + (size_t)(kb0 + 16 + row) * DH + c * 32 + hi * 16);
    }
    // ---- V fragments issued early: consumed only after softmax ----
    v16bf vf[4];
#pragma unroll
    for (int c = 0; c < 4; ++c)
      vf[c] = *(const v16bf*)(Vt + (size_t)(c * 16 + row) * NT + kb0 + hi * 16);

#pragma unroll
    for (int qt = 0; qt < 2; ++qt) {
      if (kb0 <= t0 + qt * 16 + 15) {          // wave-uniform causal skip
        v8f s0 = zero8(), s1 = zero8();
#pragma unroll
        for (int c = 0; c < 2; ++c) {
          s0 = wmma_bf16(qf[qt][c].v, kf[0][c], s0);
          s1 = wmma_bf16(qf[qt][c].v, kf[1][c], s1);
        }
        bf16* myp = plds[wave][qt];
        // online softmax: row M = r + 8*hi, key col N = row
#pragma unroll
        for (int r = 0; r < 8; ++r) {
          int M  = r + 8 * hi;
          int qi = t0 + qt * 16 + M;
          float a0 = (kb0 + row      <= qi) ? s0[r] * scale : -1e30f;
          float a1 = (kb0 + 16 + row <= qi) ? s1[r] * scale : -1e30f;
          float v = fmaxf(a0, a1);
          v = fmaxf(v, __shfl_xor(v, 1, 16));
          v = fmaxf(v, __shfl_xor(v, 2, 16));
          v = fmaxf(v, __shfl_xor(v, 4, 16));
          v = fmaxf(v, __shfl_xor(v, 8, 16));
          float mnew  = fmaxf(mrow[qt][r], v);
          float alpha = __expf(mrow[qt][r] - mnew);
          float p0 = __expf(a0 - mnew);
          float p1 = __expf(a1 - mnew);
          float ps = p0 + p1;
          ps += __shfl_xor(ps, 1, 16);
          ps += __shfl_xor(ps, 2, 16);
          ps += __shfl_xor(ps, 4, 16);
          ps += __shfl_xor(ps, 8, 16);
          lrow[qt][r] = lrow[qt][r] * alpha + ps;
          mrow[qt][r] = mnew;
#pragma unroll
          for (int c = 0; c < 4; ++c) o[qt][c][r] *= alpha;
          // stage P (C layout) into LDS row-major [M][key]
          myp[M * 32 + row]      = (bf16)p0;
          myp[M * 32 + 16 + row] = (bf16)p1;
        }
      }
    }
    __asm__ volatile("s_wait_dscnt 0" ::: "memory");

    // ---- O += P * V  (B frag contiguous thanks to V^T layout) ----
#pragma unroll
    for (int qt = 0; qt < 2; ++qt) {
      if (kb0 <= t0 + qt * 16 + 15) {
        const bf16* myp = plds[wave][qt];
        AFrag pa;   // reload P in A-matrix layout (16x32 bf16)
        pa.h[0] = *(const v8bf*)(myp + row * 32 + hi * 16);
        pa.h[1] = *(const v8bf*)(myp + row * 32 + hi * 16 + 8);
#pragma unroll
        for (int c = 0; c < 4; ++c)
          o[qt][c] = wmma_bf16(pa.v, vf[c], o[qt][c]);
      }
    }
  }

  // ---- finalize: O /= l, write attn rows (b*T+t, h*64+d) ----
  const int b = bh / NH, h = bh % NH;
#pragma unroll
  for (int qt = 0; qt < 2; ++qt) {
#pragma unroll
    for (int r = 0; r < 8; ++r) {
      float inv = 1.0f / lrow[qt][r];
      int t = t0 + qt * 16 + r + 8 * hi;
      size_t orow = (size_t)(b * NT + t) * DM + h * DH;
#pragma unroll
      for (int c = 0; c < 4; ++c)
        attn[orow + c * 16 + row] = (bf16)(o[qt][c][r] * inv);
    }
  }
}

// ---------------------------------------------------------------------------
// Kernel 4: out = attn @ W_o  (fp32 output, C layout stores)
// ---------------------------------------------------------------------------
__global__ void oproj_gemm_kernel(const bf16* __restrict__ attn,
                                  const bf16* __restrict__ wot,
                                  float* __restrict__ out) {
  const int lane = threadIdx.x & 31;
  const int wave = threadIdx.x >> 5;
  const int wid  = blockIdx.x * (blockDim.x >> 5) + wave;
  const int NG   = DM / 64;            // 16 column groups
  const int mt = wid / NG, ng = wid % NG;
  const int mbase = mt * 32, nbase = ng * 64;

  v8f acc[2][4];
#pragma unroll
  for (int q = 0; q < 2; ++q)
#pragma unroll
    for (int nb = 0; nb < 4; ++nb) acc[q][nb] = zero8();

  gemm_tile_32x64(attn, wot, DM, DM, DM, mbase, nbase, lane, acc);

  const int hi = lane >> 4, n = lane & 15;
#pragma unroll
  for (int q = 0; q < 2; ++q) {
#pragma unroll
    for (int nb = 0; nb < 4; ++nb) {
#pragma unroll
      for (int r = 0; r < 8; ++r) {
        int m = mbase + q * 16 + r + 8 * hi;
        out[(size_t)m * DM + nbase + nb * 16 + n] = acc[q][nb][r];
      }
    }
  }
}

// ---------------------------------------------------------------------------
extern "C" void kernel_launch(void* const* d_in, const int* in_sizes, int n_in,
                              void* d_out, int out_size, void* d_ws, size_t ws_size,
                              hipStream_t stream) {
  (void)in_sizes; (void)n_in; (void)out_size; (void)ws_size;
  const float* x    = (const float*)d_in[0];
  const float* wqkv = (const float*)d_in[1];
  const float* wo   = (const float*)d_in[2];
  float* out = (float*)d_out;

  char* ws = (char*)d_ws;
  size_t off = 0;
  auto alloc = [&](size_t bytes) -> char* {
    char* p = ws + off;
    off = (off + bytes + 255) & ~(size_t)255;
    return p;
  };

  bf16* xb    = (bf16*)alloc((size_t)BT * DM * 2);       // x in bf16
  bf16* wqkvt = (bf16*)alloc((size_t)3 * DM * DM * 2);   // W_qkv^T
  bf16* wot   = (bf16*)alloc((size_t)DM * DM * 2);       // W_o^T
  bf16* qbuf  = (bf16*)alloc((size_t)BT * DM * 2);       // Q (b,h,t,d)
  bf16* kbuf  = (bf16*)alloc((size_t)BT * DM * 2);       // K (b,h,t,d)
  bf16* vtbuf = (bf16*)alloc((size_t)BT * DM * 2);       // V^T (b,h,d,t)
  bf16* attn  = (bf16*)alloc((size_t)BT * DM * 2);       // attention output

  prep_kernel<<<8192, 256, 0, stream>>>(x, wqkv, wo, xb, wqkvt, wot);

  // 4096/32 m-tiles * 48 n-groups = 6144 waves -> 1536 blocks of 4 waves
  qkv_gemm_kernel<<<1536, 128, 0, stream>>>(xb, wqkvt, qbuf, kbuf, vtbuf);

  // 2*16*64 = 2048 strip waves -> 512 blocks of 4 waves
  flash_kernel<<<512, 128, 0, stream>>>(qbuf, kbuf, vtbuf, attn);

  // 128 m-tiles * 16 n-groups = 2048 waves -> 512 blocks of 4 waves
  oproj_gemm_kernel<<<512, 128, 0, stream>>>(attn, wot, out);
}
